// RPNHead_52999896432950
// MI455X (gfx1250) — matrix operand order repaired and checked by reading, not compiled, not run on hardware
//
#include <hip/hip_runtime.h>
#include <hip/hip_bf16.h>
#include <math.h>

typedef __attribute__((ext_vector_type(16))) __bf16        v16bf;
typedef __attribute__((ext_vector_type(8)))  float         v8f;
typedef __attribute__((ext_vector_type(4)))  unsigned int  u32x4;
typedef __attribute__((ext_vector_type(4)))  float         f32x4;

#define FH 64
#define FW 64
#define CCH 512
#define NBATCH 8
#define KDIM (CCH * 9)          // 4608
#define NPIX (FH * FW)          // 4096
#define MDIM (NBATCH * NPIX)    // 32768
#define NANCH 9
#define NA_IMG (NPIX * NANCH)   // 36864
#define KEEP_PRE 1000
#define KEEP_POST 300

// padded channel-last input: [8][66][68][512] bf16
#define PH 66
#define PW 68
#define XP_ELEMS ((size_t)NBATCH * PH * PW * CCH)   // 18,382,848

#define BM 128
#define BN 128
#define KT 32                   // wmma K depth
#define LDA 40                  // LDS row stride for 32-deep tiles (head gemm)
#define LDA2 72                 // LDS row stride for 64-deep tiles (conv), 144B
#define NS_CONV 72              // 9 taps * (512/64) c-blocks

union Frag { v16bf v; u32x4 q[2]; };

// CDNA5 async global->LDS copy (ASYNCcnt-tracked), b128 per lane
#define ASYNC_LD(ldsv, gv, ofs)                                              \
  asm volatile("global_load_async_to_lds_b128 %0, %1, off offset:" #ofs     \
               :: "v"(ldsv), "v"(gv) : "memory")

// ---------------------------------------------------------------- weight prep
__global__ void prep_kernel(const float* __restrict__ base_w,
                            const float* __restrict__ cls_w,
                            const float* __restrict__ reg_w,
                            const float* __restrict__ cls_b,
                            const float* __restrict__ reg_b,
                            __bf16* __restrict__ Wr,   // [512][9][512] (tap-major)
                            __bf16* __restrict__ Hb,   // [64][512]
                            float*  __restrict__ hbias)
{
  int i = blockIdx.x * blockDim.x + threadIdx.x;
  if (i < CCH * KDIM) {
    int o   = i / KDIM;
    int rem = i - o * KDIM;          // r*512 + c
    int r   = rem >> 9, c = rem & 511;
    Wr[i] = (__bf16)base_w[o * KDIM + c * 9 + r];   // OIHW -> O,(ky,kx),C
  }
  if (i < 64 * CCH) {
    int n = i >> 9, c = i & 511;
    float v = 0.f;
    if (n < 9)       v = cls_w[n * CCH + c];
    else if (n < 45) v = reg_w[(n - 9) * CCH + c];
    Hb[i] = (__bf16)v;
  }
  if (i < 64) {
    float v = 0.f;
    if (i < 9)       v = cls_b[i];
    else if (i < 45) v = reg_b[i - 9];
    hbias[i] = v;
  }
}

// ---------------------------------------------------------------- pad zeroing
__global__ void pad_zero(__bf16* __restrict__ Xp)
{
  size_t i = (size_t)blockIdx.x * blockDim.x + threadIdx.x;   // 8 bf16 per thread
  if (i < XP_ELEMS / 8) ((u32x4*)Xp)[i] = (u32x4){0u, 0u, 0u, 0u};
}

// ----------------------------------- NCHW fp32 -> padded channel-last bf16
__global__ __launch_bounds__(256) void cvt_pad(const float* __restrict__ fm,
                                               __bf16* __restrict__ Xp)
{
  __shared__ float t[64][65];
  const int by = blockIdx.x;
  const int b  = by >> 6;
  const int y  = by & 63;
  const int tid = threadIdx.x;
  const int ci  = tid >> 2, x0 = (tid & 3) * 16;   // load role
  const int xo  = tid >> 2, c0l = (tid & 3) * 16;  // store role

  for (int c0 = 0; c0 < CCH; c0 += 64) {
    const float* src = fm + ((((size_t)b * CCH + c0 + ci) * FH + y) * FW + x0);
    f32x4 r0 = *(const f32x4*)(src + 0);
    f32x4 r1 = *(const f32x4*)(src + 4);
    f32x4 r2 = *(const f32x4*)(src + 8);
    f32x4 r3 = *(const f32x4*)(src + 12);
    #pragma unroll
    for (int j = 0; j < 4; ++j) t[ci][x0 + j]      = r0[j];
    #pragma unroll
    for (int j = 0; j < 4; ++j) t[ci][x0 + 4 + j]  = r1[j];
    #pragma unroll
    for (int j = 0; j < 4; ++j) t[ci][x0 + 8 + j]  = r2[j];
    #pragma unroll
    for (int j = 0; j < 4; ++j) t[ci][x0 + 12 + j] = r3[j];
    __syncthreads();
    __bf16* dst = Xp + (((size_t)b * PH + y + 1) * PW + (xo + 1)) * CCH + c0 + c0l;
    #pragma unroll
    for (int j = 0; j < 16; ++j) dst[j] = (__bf16)t[c0l + j][xo];
    __syncthreads();
  }
}

// ------------------------------------------------- 3x3 conv as implicit GEMM
// K ordered (ky,kx)-major, c-minor; padded channel-last input => every tile
// row is a contiguous aligned 128B run loaded with async b128 -> LDS pingpong.
// KT=64 per tile (two k32 wmma sub-steps) to amortize barriers/async issues.
__global__ __launch_bounds__(256) void conv3x3_wmma(
    const __bf16* __restrict__ Xp,   // [8][66][68][512]
    const __bf16* __restrict__ W,    // [512][9][512]
    const float*  __restrict__ bias, // [512]
    __bf16* __restrict__ Y)          // [32768][512] post bias+relu
{
  __shared__ __bf16 As[2][BM * LDA2];
  __shared__ __bf16 Bs[2][BM * LDA2];
  const int tid   = threadIdx.x;
  const int lane  = tid & 31;
  const int wave  = tid >> 5;
  const int wm    = wave >> 1;           // 0..3
  const int wn    = wave & 1;            // 0..1
  const int bm0   = blockIdx.x * BM;
  const int bn0   = blockIdx.y * BN;
  const int lhalf = lane >> 4;
  const int lrow  = lane & 15;

  v8f acc[2][4] = {};

  // copy role: row = tid>>1, 64B half-row = tid&1 (4 async b128 each)
  const int ar = tid >> 1, ah = tid & 1;
  const int am = bm0 + ar;
  const int ab = am >> 12, apix = am & (NPIX - 1);
  const int ay = apix >> 6, ax = apix & 63;
  const unsigned long long baseA = (unsigned long long)Xp +
      2ull * ((((size_t)(ab * PH + ay)) * PW + ax) * CCH) + (unsigned)(ah * 64);
  const unsigned long long baseB = (unsigned long long)W +
      2ull * ((size_t)(bn0 + ar) * KDIM) + (unsigned)(ah * 64);
  const unsigned ldsA0 = (unsigned)(unsigned long long)(void*)&As[0][ar * LDA2] + ah * 64;
  const unsigned ldsB0 = (unsigned)(unsigned long long)(void*)&Bs[0][ar * LDA2] + ah * 64;
  const unsigned bufoff = BM * LDA2 * 2;  // 18432 B

  auto issue = [&](int s) {
    int r  = s >> 3;                     // tap 0..8
    int cb = s & 7;                      // 64-wide c block
    int ky = r / 3, kx = r - ky * 3;
    unsigned offA = (unsigned)((ky * PW + kx) * (CCH * 2) + cb * 128);
    unsigned offB = (unsigned)(r * (CCH * 2) + cb * 128);
    unsigned long long gA = baseA + offA;
    unsigned long long gB = baseB + offB;
    unsigned lA = ldsA0 + (unsigned)((s & 1) * bufoff);
    unsigned lB = ldsB0 + (unsigned)((s & 1) * bufoff);
    ASYNC_LD(lA, gA, 0);
    ASYNC_LD(lA, gA, 16);
    ASYNC_LD(lA, gA, 32);
    ASYNC_LD(lA, gA, 48);
    ASYNC_LD(lB, gB, 0);
    ASYNC_LD(lB, gB, 16);
    ASYNC_LD(lB, gB, 32);
    ASYNC_LD(lB, gB, 48);
  };

  issue(0);
  #pragma unroll 1
  for (int s = 0; s < NS_CONV; ++s) {
    if (s + 1 < NS_CONV) {
      issue(s + 1);
      asm volatile("s_wait_asynccnt 0x8" ::: "memory");  // tile s landed
    } else {
      asm volatile("s_wait_asynccnt 0x0" ::: "memory");
    }
    __syncthreads();

    const __bf16* Ab = As[s & 1];
    const __bf16* Bb = Bs[s & 1];
    #pragma unroll
    for (int ks = 0; ks < 2; ++ks) {     // two k32 sub-steps per tile
      Frag a[2], bb[4];
      #pragma unroll
      for (int mi = 0; mi < 2; ++mi) {   // A 16x32 frag
        int row = wm * 32 + mi * 16 + lrow;
        a[mi].q[0] = *(const u32x4*)&Ab[row * LDA2 + ks * 32 + lhalf * 8];
        a[mi].q[1] = *(const u32x4*)&Ab[row * LDA2 + ks * 32 + lhalf * 8 + 16];
      }
      #pragma unroll
      for (int ni = 0; ni < 4; ++ni) {   // B 32x16 frag
        int col = wn * 64 + ni * 16 + lrow;
        bb[ni].q[0] = *(const u32x4*)&Bb[col * LDA2 + ks * 32 + lhalf * 8];
        bb[ni].q[1] = *(const u32x4*)&Bb[col * LDA2 + ks * 32 + lhalf * 8 + 16];
      }
      #pragma unroll
      for (int mi = 0; mi < 2; ++mi)
        #pragma unroll
        for (int ni = 0; ni < 4; ++ni)
          acc[mi][ni] = __builtin_amdgcn_wmma_f32_16x16x32_bf16(
              false, a[mi].v, false, bb[ni].v, (short)0, acc[mi][ni], false, false);
    }
    __syncthreads();                     // protect buf before next overwrite
  }

  #pragma unroll
  for (int mi = 0; mi < 2; ++mi)
    #pragma unroll
    for (int ni = 0; ni < 4; ++ni) {
      int N = bn0 + wn * 64 + ni * 16 + lrow;
      float bv = bias[N];
      #pragma unroll
      for (int r = 0; r < 8; ++r) {
        int M = bm0 + wm * 32 + mi * 16 + lhalf * 8 + r;
        float val = acc[mi][ni][r] + bv;
        val = val > 0.f ? val : 0.f;
        Y[(size_t)M * CCH + N] = (__bf16)val;
      }
    }
}

// ------------------------------------------------- fused cls/reg 1x1 head GEMM
__global__ __launch_bounds__(256) void head_wmma(
    const __bf16* __restrict__ X,    // [32768][512]
    const __bf16* __restrict__ Wh,   // [64][512]
    const float*  __restrict__ hbias,
    float* __restrict__ O)           // [32768][64]
{
  __shared__ __bf16 As[BM * LDA];
  __shared__ __bf16 Bs[64 * LDA];
  const int tid   = threadIdx.x;
  const int lane  = tid & 31;
  const int wave  = tid >> 5;
  const int wm    = wave >> 1;
  const int wn    = wave & 1;
  const int bm0   = blockIdx.x * BM;
  const int lhalf = lane >> 4;
  const int lrow  = lane & 15;

  v8f acc[2][2] = {};
  const int ar = tid >> 1, akoff = (tid & 1) * 16;
  const int bn = tid >> 2, bkoff = (tid & 3) * 8;

  for (int k0 = 0; k0 < CCH; k0 += KT) {
    {
      const __bf16* src = X + (size_t)(bm0 + ar) * CCH + k0 + akoff;
      *(u32x4*)&As[ar * LDA + akoff]     = *(const u32x4*)(src);
      *(u32x4*)&As[ar * LDA + akoff + 8] = *(const u32x4*)(src + 8);
    }
    {
      const __bf16* src = Wh + (size_t)bn * CCH + k0 + bkoff;
      *(u32x4*)&Bs[bn * LDA + bkoff] = *(const u32x4*)(src);
    }
    __syncthreads();

    Frag a[2], bb[2];
    #pragma unroll
    for (int mi = 0; mi < 2; ++mi) {
      int row = wm * 32 + mi * 16 + lrow;
      int kb  = lhalf * 8;
      a[mi].q[0] = *(const u32x4*)&As[row * LDA + kb];
      a[mi].q[1] = *(const u32x4*)&As[row * LDA + kb + 16];
    }
    #pragma unroll
    for (int ni = 0; ni < 2; ++ni) {
      int col = wn * 32 + ni * 16 + lrow;
      int kb  = lhalf * 8;
      bb[ni].q[0] = *(const u32x4*)&Bs[col * LDA + kb];
      bb[ni].q[1] = *(const u32x4*)&Bs[col * LDA + kb + 16];
    }
    #pragma unroll
    for (int mi = 0; mi < 2; ++mi)
      #pragma unroll
      for (int ni = 0; ni < 2; ++ni)
        acc[mi][ni] = __builtin_amdgcn_wmma_f32_16x16x32_bf16(
            false, a[mi].v, false, bb[ni].v, (short)0, acc[mi][ni], false, false);
    __syncthreads();
  }

  #pragma unroll
  for (int mi = 0; mi < 2; ++mi)
    #pragma unroll
    for (int ni = 0; ni < 2; ++ni) {
      int N = wn * 32 + ni * 16 + lrow;
      float bv = hbias[N];
      #pragma unroll
      for (int r = 0; r < 8; ++r) {
        int M = bm0 + wm * 32 + mi * 16 + lhalf * 8 + r;
        O[(size_t)M * 64 + N] = acc[mi][ni][r] + bv;
      }
    }
}

// ------------------------------------------------- anchors + sigmoid + decode
__global__ void decode_kernel(const float* __restrict__ P,   // [32768][64]
                              const int* __restrict__ img_h_p,
                              const int* __restrict__ img_w_p,
                              float* __restrict__ boxes,     // [8][36864][4]
                              float* __restrict__ scores)    // [8][36864]
{
  int gid = blockIdx.x * blockDim.x + threadIdx.x;
  if (gid >= NBATCH * NA_IMG) return;
  int b   = gid / NA_IMG;
  int t   = gid - b * NA_IMG;
  int pix = t / NANCH;
  int a   = t - pix * NANCH;
  int y   = pix >> 6, x = pix & 63;
  size_t m = (size_t)b * NPIX + pix;
  const float* row = P + m * 64;

  float score = 1.f / (1.f + __expf(-row[a]));
  float dx = row[9 + a * 4 + 0];
  float dy = row[9 + a * 4 + 1];
  float dw = row[9 + a * 4 + 2];
  float dh = row[9 + a * 4 + 3];

  const float sizes[3]  = {128.f, 256.f, 512.f};
  const float ratios[3] = {0.5f, 1.f, 2.f};
  float sz = sizes[a % 3];
  float hr = sqrtf(ratios[a / 3]);
  float aw = (1.f / hr) * sz, ah = hr * sz;

  float img_w = (float)img_w_p[0], img_h = (float)img_h_p[0];
  float acx = x * (img_w / FW), acy = y * (img_h / FH);
  float pcx = dx * aw + acx, pcy = dy * ah + acy;
  float pw  = __expf(dw) * aw, ph = __expf(dh) * ah;

  float x1 = fminf(fmaxf(pcx - 0.5f * pw, 0.f), img_w);
  float y1 = fminf(fmaxf(pcy - 0.5f * ph, 0.f), img_h);
  float x2 = fminf(fmaxf(pcx + 0.5f * pw, 0.f), img_w);
  float y2 = fminf(fmaxf(pcy + 0.5f * ph, 0.f), img_h);

  size_t o = (size_t)gid * 4;
  boxes[o + 0] = x1; boxes[o + 1] = y1; boxes[o + 2] = x2; boxes[o + 3] = y2;
  scores[gid] = score;
}

// ------------------------------------------------- per-image top-k + NMS
__device__ inline unsigned long long sel_key(float s, int idx)
{
  unsigned u = __float_as_uint(s);
  u = (u & 0x80000000u) ? ~u : (u | 0x80000000u);
  return (((unsigned long long)u) << 32) | (unsigned)(0xFFFFFFFFu - (unsigned)idx);
}

__global__ __launch_bounds__(1024) void topk_nms_kernel(
    float* __restrict__ scores, const float* __restrict__ boxes,
    float* __restrict__ out)
{
  const int b = blockIdx.x;
  float* sc = scores + (size_t)b * NA_IMG;
  const float* bx = boxes + (size_t)b * NA_IMG * 4;
  const int tid = threadIdx.x;
  const int lane = tid & 31;
  const int wv = tid >> 5;

  __shared__ unsigned long long red[32];
  __shared__ unsigned long long winner;
  __shared__ float sx1[KEEP_PRE], sy1[KEEP_PRE], sx2[KEEP_PRE], sy2[KEEP_PRE], ssc[KEEP_PRE];
  __shared__ int keep_s[1024];
  __shared__ int scan[1024];

  const int c0 = tid * 36;
  unsigned long long best = 0ull;
  for (int j = 0; j < 36; ++j) {
    unsigned long long k = sel_key(sc[c0 + j], c0 + j);
    best = k > best ? k : best;
  }

  for (int s = 0; s < KEEP_PRE; ++s) {
    unsigned long long k = best;
    #pragma unroll
    for (int off = 16; off; off >>= 1) {
      unsigned long long o = __shfl_xor(k, off, 32);
      k = o > k ? o : k;
    }
    if (lane == 0) red[wv] = k;
    __syncthreads();
    if (tid < 32) {
      unsigned long long k2 = red[tid];
      #pragma unroll
      for (int off = 16; off; off >>= 1) {
        unsigned long long o = __shfl_xor(k2, off, 32);
        k2 = o > k2 ? o : k2;
      }
      if (tid == 0) winner = k2;
    }
    __syncthreads();
    int widx = (int)(0xFFFFFFFFu - (unsigned)(winner & 0xFFFFFFFFull));
    if (widx >= c0 && widx < c0 + 36) {
      ssc[s] = sc[widx];
      sx1[s] = bx[(size_t)widx * 4 + 0];
      sy1[s] = bx[(size_t)widx * 4 + 1];
      sx2[s] = bx[(size_t)widx * 4 + 2];
      sy2[s] = bx[(size_t)widx * 4 + 3];
      sc[widx] = -__builtin_inff();
      unsigned long long m = 0ull;
      for (int j = 0; j < 36; ++j) {
        unsigned long long kk = sel_key(sc[c0 + j], c0 + j);
        m = kk > m ? kk : m;
      }
      best = m;
    }
    __syncthreads();
  }

  float mx1 = 0, my1 = 0, mx2 = 0, my2 = 0, msc = 0, marea = 0;
  if (tid < KEEP_PRE) {
    mx1 = sx1[tid]; my1 = sy1[tid]; mx2 = sx2[tid]; my2 = sy2[tid]; msc = ssc[tid];
    marea = (mx2 - mx1) * (my2 - my1);
  }
  keep_s[tid] = (tid < KEEP_PRE) ? 1 : 0;
  __syncthreads();
  for (int i = 0; i < KEEP_PRE - 1; ++i) {
    if (keep_s[i] && tid > i && tid < KEEP_PRE && keep_s[tid]) {
      float ix1 = sx1[i], iy1 = sy1[i], ix2 = sx2[i], iy2 = sy2[i];
      float ia = (ix2 - ix1) * (iy2 - iy1);
      float lx = fmaxf(ix1, mx1), ly = fmaxf(iy1, my1);
      float rx = fminf(ix2, mx2), ry = fminf(iy2, my2);
      float w = fmaxf(rx - lx, 0.f), h = fmaxf(ry - ly, 0.f);
      float inter = w * h;
      float iou = inter / (ia + marea - inter + 1e-9f);
      if (iou > 0.7f) keep_s[tid] = 0;
    }
    __syncthreads();
  }

  scan[tid] = keep_s[tid];
  __syncthreads();
  for (int off = 1; off < 1024; off <<= 1) {
    int v = scan[tid];
    int add = (tid >= off) ? scan[tid - off] : 0;
    __syncthreads();
    scan[tid] = v + add;
    __syncthreads();
  }
  int pos = scan[tid] - keep_s[tid];
  float* ob = out + (size_t)b * KEEP_POST * 5;
  for (int j = tid; j < KEEP_POST * 5; j += 1024) ob[j] = 0.f;
  __syncthreads();
  if (tid < KEEP_PRE && keep_s[tid] && pos < KEEP_POST) {
    float* o = ob + pos * 5;
    o[0] = mx1; o[1] = my1; o[2] = mx2; o[3] = my2; o[4] = msc;
  }
}

// ---------------------------------------------------------------- launcher
extern "C" void kernel_launch(void* const* d_in, const int* in_sizes, int n_in,
                              void* d_out, int out_size, void* d_ws, size_t ws_size,
                              hipStream_t stream)
{
  (void)in_sizes; (void)n_in; (void)out_size; (void)ws_size;
  const float* fmaps  = (const float*)d_in[0];
  const float* base_w = (const float*)d_in[1];
  const float* base_b = (const float*)d_in[2];
  const float* cls_w  = (const float*)d_in[3];
  const float* cls_b  = (const float*)d_in[4];
  const float* reg_w  = (const float*)d_in[5];
  const float* reg_b  = (const float*)d_in[6];
  const int*   img_h  = (const int*)d_in[7];
  const int*   img_w  = (const int*)d_in[8];

  char* ws = (char*)d_ws;
  size_t o = 0;
  __bf16* Xp     = (__bf16*)(ws + o); o += XP_ELEMS * 2;                  // 35.1 MiB
  __bf16* Wr     = (__bf16*)(ws + o); o += (size_t)CCH * KDIM * 2;        // 4.5 MiB
  __bf16* Hb     = (__bf16*)(ws + o); o += (size_t)64 * CCH * 2;
  float*  hbias  = (float*) (ws + o); o += 64 * sizeof(float);
  o = (o + 255) & ~(size_t)255;
  __bf16* xact   = (__bf16*)(ws + o); o += (size_t)MDIM * CCH * 2;        // 32 MiB
  float*  pred   = (float*) (ws + o); o += (size_t)MDIM * 64 * 4;         // 8 MiB
  float*  boxes  = (float*) (ws + o); o += (size_t)NBATCH * NA_IMG * 4 * 4;
  float*  scores = (float*) (ws + o); o += (size_t)NBATCH * NA_IMG * 4;

  prep_kernel<<<(CCH * KDIM + 255) / 256, 256, 0, stream>>>(
      base_w, cls_w, reg_w, cls_b, reg_b, Wr, Hb, hbias);
  pad_zero<<<(int)((XP_ELEMS / 8 + 255) / 256), 256, 0, stream>>>(Xp);
  cvt_pad<<<NBATCH * FH, 256, 0, stream>>>(fmaps, Xp);
  conv3x3_wmma<<<dim3(MDIM / BM, CCH / BN), 256, 0, stream>>>(
      Xp, Wr, base_b, xact);
  head_wmma<<<MDIM / BM, 256, 0, stream>>>(xact, Hb, hbias, pred);
  decode_kernel<<<(NBATCH * NA_IMG + 255) / 256, 256, 0, stream>>>(
      pred, img_h, img_w, boxes, scores);
  topk_nms_kernel<<<NBATCH, 1024, 0, stream>>>(scores, boxes, (float*)d_out);
}
